// TripleGATLayerV2_44220983279642
// MI455X (gfx1250) — matrix operand order repaired and checked by reading, not compiled, not run on hardware
//
#include <hip/hip_runtime.h>
#include <hip/hip_bf16.h>

// ---------------------------------------------------------------------------
// Triple-GAT layer for MI455X (gfx1250, wave32, WMMA).
// B=16, N=256, IN_F=128, OUT_F=64, HEADS=4 (12 attention channels).
//
// Stage A: h = x @ W^T via v_wmma_f32_16x16x32_f16 (f32->f16 inline convert),
//          plus e_src/e_dst projections (A staged in LDS); h stored f16.
// Stage B: per (b, 16 query rows): masked leaky-relu scores, wave-shuffle
//          softmax, att(16x256 f16) @ h(256x64 f16) via 8x WMMA per head.
//          B-fragments (h^T) register-cached across the 12-head loop;
//          per-head 1/rowsum folded post-WMMA; mean over 12 heads.
// ---------------------------------------------------------------------------

typedef __attribute__((ext_vector_type(16))) _Float16 v16h;
typedef __attribute__((ext_vector_type(8)))  _Float16 v8h;
typedef __attribute__((ext_vector_type(8)))  float    v8f;

union V16 { v16h v; v8h h[2]; };

__device__ inline v8h cvt8(float4 a, float4 b) {
  v8h r;
  r[0] = (_Float16)a.x; r[1] = (_Float16)a.y; r[2] = (_Float16)a.z; r[3] = (_Float16)a.w;
  r[4] = (_Float16)b.x; r[5] = (_Float16)b.y; r[6] = (_Float16)b.z; r[7] = (_Float16)b.w;
  return r;
}

#define NEGV (-9.0e15f)

// ============================ Kernel A =====================================
// grid: 64 blocks x 128 threads (4 waves). One wave = one 16-row tile of the
// flat (4096 x 128) input. Output: h f16 (4096x64), e_src/e_dst (4096x12).
__global__ __launch_bounds__(128) void gat_h_proj_kernel(
    const float* __restrict__ x,     // (4096,128)
    const float* __restrict__ W,     // (64,128)
    const float* __restrict__ A,     // (12,128) [A1 | A2]
    _Float16* __restrict__ hbuf,     // (4096,64) f16
    float* __restrict__ es,          // (4096,12)
    float* __restrict__ ed)          // (4096,12)
{
  __shared__ float lds_h[4][16][64];
  __shared__ float lds_A[12][128];

  const int tid  = threadIdx.x;
  const int w    = tid >> 5;
  const int lane = tid & 31;
  const int lo   = lane & 15;
  const int hi   = lane >> 4;
  const int tile = blockIdx.x * 4 + w;          // 0..255

  // stage A (12x128 f32) into LDS for the projection phase
  for (int idx = tid; idx < 12 * 128; idx += 128)
    ((float*)lds_A)[idx] = A[idx];

  const float* xrow = x + (size_t)(tile * 16 + lo) * 128;

  const v8f vzero = {0.f,0.f,0.f,0.f,0.f,0.f,0.f,0.f};
  v8f acc[4];
#pragma unroll
  for (int t = 0; t < 4; ++t) acc[t] = vzero;

#pragma unroll
  for (int ks = 0; ks < 4; ++ks) {
    const int k0 = ks * 32;
    // A fragment: 16x32 f16, ISA wave32 layout (lane halves split K).
    V16 afr;
    {
      const float4* p0 = (const float4*)(xrow + k0 + hi * 8);
      const float4* p1 = (const float4*)(xrow + k0 + 16 + hi * 8);
      afr.h[0] = cvt8(p0[0], p0[1]);
      afr.h[1] = cvt8(p1[0], p1[1]);
    }
#pragma unroll
    for (int t = 0; t < 4; ++t) {
      // B fragment: column o = t*16+lo (= row of W), K split like A (lane halves).
      const float4* q = (const float4*)(W + (size_t)(t * 16 + lo) * 128 + k0 + hi * 8);
      V16 bfr;
      bfr.h[0] = cvt8(q[0], q[1]);
      bfr.h[1] = cvt8(q[4], q[5]);
      acc[t] = __builtin_amdgcn_wmma_f32_16x16x32_f16(
          false, afr.v, false, bfr.v, (short)0, acc[t], false, false);
    }
  }

  // Spill tile to LDS (for projections) + f16 copy to scratch.
#pragma unroll
  for (int t = 0; t < 4; ++t) {
#pragma unroll
    for (int r = 0; r < 8; ++r) {
      const int row = r + 8 * hi;
      const int col = t * 16 + lo;
      const float v = acc[t][r];
      lds_h[w][row][col] = v;
      hbuf[(size_t)(tile * 16 + row) * 64 + col] = (_Float16)v;
    }
  }
  __syncthreads();

  // e_src / e_dst: 16 rows x 12 channels, 64-length dots against A1/A2.
  for (int idx = lane; idx < 16 * 12; idx += 32) {
    const int row = idx / 12;
    const int hh  = idx % 12;
    float s1 = 0.f, s2 = 0.f;
#pragma unroll 8
    for (int d = 0; d < 64; ++d) {
      const float hv = lds_h[w][row][d];
      s1 = fmaf(hv, lds_A[hh][d],      s1);
      s2 = fmaf(hv, lds_A[hh][64 + d], s2);
    }
    const int gi = tile * 16 + row;
    es[(size_t)gi * 12 + hh] = s1;
    ed[(size_t)gi * 12 + hh] = s2;
  }
}

// ============================ Kernel B =====================================
// grid: 256 blocks (b = blk>>4, i0 = (blk&15)*16) x 128 threads (4 waves).
__global__ __launch_bounds__(128) void gat_attn_kernel(
    const _Float16* __restrict__ hbuf,  // (B,256,64) f16
    const float* __restrict__ es_g,     // (B,256,12)
    const float* __restrict__ ed_g,     // (B,256,12)
    const int* __restrict__ adj,
    const int* __restrict__ adj_in,
    const int* __restrict__ adj_out,
    const float* __restrict__ ab,       // (12,)
    float* __restrict__ out)            // (B,256,64)
{
  __shared__ __align__(16) _Float16 lds_ht[64][264];  // h^T, padded pitch
  __shared__ __align__(16) _Float16 lds_att[16][256]; // unnormalized exp
  __shared__ float lds_ed[256][12];
  __shared__ float lds_es[16][12];
  __shared__ float lds_rs[16];                        // 1/rowsum per head
  __shared__ float lds_ab[12];

  const int tid  = threadIdx.x;
  const int w    = tid >> 5;
  const int lane = tid & 31;
  const int lo   = lane & 15;
  const int hi   = lane >> 4;
  const int b    = blockIdx.x >> 4;
  const int i0   = (blockIdx.x & 15) * 16;

  // ---- stage values into LDS ----
  const _Float16* hb = hbuf + (size_t)b * 256 * 64;
  for (int c = tid; c < 2048; c += 128) {             // 256 rows x 8 chunks
    const int j  = c >> 3;
    const int d0 = (c & 7) * 8;
    const v8h hv = *(const v8h*)(hb + (size_t)j * 64 + d0);
#pragma unroll
    for (int u = 0; u < 8; ++u) lds_ht[d0 + u][j] = hv[u];  // transpose
  }
  for (int idx = tid; idx < 3072; idx += 128)
    lds_ed[idx / 12][idx % 12] = ed_g[(size_t)b * 3072 + idx];
  for (int idx = tid; idx < 192; idx += 128)
    lds_es[idx / 12][idx % 12] = es_g[(size_t)(b * 256 + i0) * 12 + idx];
  if (tid < 12) lds_ab[tid] = ab[tid];
  __syncthreads();

  // ---- register-cache the B fragments (h^T slab for this wave's 16 cols) --
  // lds_ht is invariant across the 12-head loop: 8 K-steps x v16h = 64 VGPRs.
  V16 bfr[8];
#pragma unroll
  for (int kt = 0; kt < 8; ++kt) {
    const _Float16* bp = &lds_ht[w * 16 + lo][kt * 32 + hi * 8];
    bfr[kt].h[0] = *(const v8h*)bp;
    bfr[kt].h[1] = *(const v8h*)(bp + 16);
  }

  const v8f vzero = {0.f,0.f,0.f,0.f,0.f,0.f,0.f,0.f};
  v8f acc = vzero;

  for (int hh = 0; hh < 12; ++hh) {
    const int* adjsel = (hh < 4) ? adj : ((hh < 8) ? adj_in : adj_out);
    const int* adjb   = adjsel + (size_t)b * 65536;

    // ---- scores + softmax: wave w owns query rows 4w..4w+3 ----
#pragma unroll
    for (int rr = 0; rr < 4; ++rr) {
      const int r = w * 4 + rr;
      const int i = i0 + r;
      const float base = lds_es[r][hh] + lds_ab[hh];
      const int* arow = adjb + (size_t)i * 256;
      if (rr < 3)
        __builtin_prefetch(adjb + (size_t)(i + 1) * 256 + lane * 8, 0, 0);

      float sc[8];
      float mx = -3.4e38f;
#pragma unroll
      for (int u = 0; u < 8; ++u) {
        const int j = lane + 32 * u;
        float v = base + lds_ed[j][hh];
        v = (v > 0.f) ? v : 0.01f * v;           // leaky_relu(0.01)
        v = (arow[j] > 0) ? v : NEGV;            // adjacency mask
        sc[u] = v;
        mx = fmaxf(mx, v);
      }
#pragma unroll
      for (int m = 16; m > 0; m >>= 1) mx = fmaxf(mx, __shfl_xor(mx, m, 32));
      float s = 0.f;
#pragma unroll
      for (int u = 0; u < 8; ++u) {
        const float p = __expf(sc[u] - mx);
        s += p;
        lds_att[r][lane + 32 * u] = (_Float16)p;
      }
#pragma unroll
      for (int m = 16; m > 0; m >>= 1) s += __shfl_xor(s, m, 32);
      if (lane == 0) lds_rs[r] = 1.0f / s;
    }
    __syncthreads();

    // ---- att(16x256) @ h(256x64): wave w owns output cols 16w..16w+15 ----
    v8f c = vzero;
#pragma unroll
    for (int kt = 0; kt < 8; ++kt) {
      const int k0 = kt * 32;
      V16 afr;
      afr.h[0] = *(const v8h*)&lds_att[lo][k0 + hi * 8];
      afr.h[1] = *(const v8h*)&lds_att[lo][k0 + 16 + hi * 8];
      c = __builtin_amdgcn_wmma_f32_16x16x32_f16(
          false, afr.v, false, bfr[kt].v, (short)0, c, false, false);
    }
    // fold per-row softmax normalization, accumulate across heads in regs
#pragma unroll
    for (int r = 0; r < 8; ++r) acc[r] += c[r] * lds_rs[r + 8 * hi];
    __syncthreads();   // before next head rewrites lds_att / lds_rs
  }

  // mean over 12 heads, store
  const float scale = 1.0f / 12.0f;
#pragma unroll
  for (int r = 0; r < 8; ++r) {
    const int row = r + 8 * hi;
    const int col = w * 16 + lo;
    out[((size_t)b * 256 + i0 + row) * 64 + col] = acc[r] * scale;
  }
}

// ============================ Launch =======================================
extern "C" void kernel_launch(void* const* d_in, const int* in_sizes, int n_in,
                              void* d_out, int out_size, void* d_ws, size_t ws_size,
                              hipStream_t stream) {
  const float* x      = (const float*)d_in[0];   // (16,256,128)
  const int*   adj    = (const int*)d_in[1];     // (16,256,256)
  const int*   adj_in = (const int*)d_in[2];
  const int*   adj_out= (const int*)d_in[3];
  const float* W      = (const float*)d_in[4];   // (64,128)
  const float* A      = (const float*)d_in[5];   // (12,128)
  const float* ab     = (const float*)d_in[6];   // (12,)
  float* out = (float*)d_out;                    // (16,256,64)

  char* ws = (char*)d_ws;
  _Float16* hbuf = (_Float16*)ws;                          // 4096*64*2  = 512KB
  float*    es   = (float*)(ws + (size_t)4096 * 64 * 2);   // 4096*12*4  = 192KB
  float*    ed   = (float*)(ws + (size_t)4096 * 64 * 2
                               + (size_t)4096 * 12 * 4);   // 192KB

  gat_h_proj_kernel<<<64, 128, 0, stream>>>(x, W, A, hbuf, es, ed);
  gat_attn_kernel<<<256, 128, 0, stream>>>(hbuf, es, ed, adj, adj_in, adj_out, ab, out);
}
